// SimpleRGATLayer_4071628996917
// MI455X (gfx1250) — matrix operand order repaired and checked by reading, not compiled, not run on hardware
//
#include <hip/hip_runtime.h>
#include <hip/hip_bf16.h>
#include <math.h>

#define N_NODES 50000
#define N_EDGES 1600000
#define IN_DIM  256
#define OUT_DIM 32
#define HEADS   4
#define M_OUT   (OUT_DIM * HEADS)   /* 128 */

typedef __attribute__((ext_vector_type(16))) __bf16 v16bf;
typedef __attribute__((ext_vector_type(8)))  float  v8f;

// ---------------------------------------------------------------------------
// init: zero out, denom; emax = -inf
// ---------------------------------------------------------------------------
__global__ __launch_bounds__(256)
void k_init(float* __restrict__ out, float* __restrict__ emax, float* __restrict__ denom) {
    long long i = (long long)blockIdx.x * blockDim.x + threadIdx.x;
    if (i < (long long)N_NODES * M_OUT) out[i] = 0.0f;
    if (i < (long long)N_NODES * HEADS) {
        ((unsigned int*)emax)[i] = 0xFF800000u;  // -inf bits
        denom[i] = 0.0f;
    }
}

// ---------------------------------------------------------------------------
// GEMM: h[N,128] = x[N,256] @ W[128,256]^T   via v_wmma_f32_16x16x32_bf16
// Block = 256 threads (8 waves). Block tile: 128 rows x 128 cols, K-chunks of 64
// staged in LDS as bf16. Each wave owns a 16-row stripe and all 8 N-tiles.
// ---------------------------------------------------------------------------
__global__ __launch_bounds__(256)
void k_gemm_wmma(const float* __restrict__ x, const float* __restrict__ W,
                 float* __restrict__ h) {
    __shared__ __bf16 sx[128 * 64];   // 16 KB
    __shared__ __bf16 sw[128 * 64];   // 16 KB

    const int tid    = threadIdx.x;
    const int lane   = tid & 31;
    const int wv     = tid >> 5;        // wave 0..7 -> M stripe
    const int laneHi = lane >> 4;       // 0/1
    const int laneLo = lane & 15;
    const int rowBase = blockIdx.x * 128;

    v8f acc[8] = {};

    for (int kc = 0; kc < IN_DIM; kc += 64) {
        // ---- stage x tile (128x64) f32 -> bf16 ----
        #pragma unroll
        for (int i = 0; i < 8; ++i) {
            int v4 = tid + i * 256;          // vec4 index 0..2047
            int r  = v4 >> 4;                // 16 float4 per 64-wide row
            int c4 = (v4 & 15) << 2;
            float4 val = make_float4(0.f, 0.f, 0.f, 0.f);
            int gr = rowBase + r;
            if (gr < N_NODES)
                val = *reinterpret_cast<const float4*>(x + (long long)gr * IN_DIM + kc + c4);
            sx[r * 64 + c4 + 0] = (__bf16)val.x;
            sx[r * 64 + c4 + 1] = (__bf16)val.y;
            sx[r * 64 + c4 + 2] = (__bf16)val.z;
            sx[r * 64 + c4 + 3] = (__bf16)val.w;
        }
        // ---- stage W tile (128x64) f32 -> bf16 ----
        #pragma unroll
        for (int i = 0; i < 8; ++i) {
            int v4 = tid + i * 256;
            int r  = v4 >> 4;
            int c4 = (v4 & 15) << 2;
            float4 val = *reinterpret_cast<const float4*>(W + (long long)r * IN_DIM + kc + c4);
            sw[r * 64 + c4 + 0] = (__bf16)val.x;
            sw[r * 64 + c4 + 1] = (__bf16)val.y;
            sw[r * 64 + c4 + 2] = (__bf16)val.z;
            sw[r * 64 + c4 + 3] = (__bf16)val.w;
        }
        __syncthreads();

        #pragma unroll
        for (int ks = 0; ks < 64; ks += 32) {
            // A fragment: 16x32 bf16, layout:
            //   elem e -> K = laneHi*16 + e, row M = laneLo
            v16bf a;
            #pragma unroll
            for (int e = 0; e < 16; ++e) {
                int K = (laneHi << 4) + e;
                a[e] = sx[(wv * 16 + laneLo) * 64 + ks + K];
            }
            #pragma unroll
            for (int nt = 0; nt < 8; ++nt) {
                // B fragment: 32x16 bf16 (B[k][n] = W[n][k]), layout:
                //   elem e -> K = laneHi*16 + e, col N = laneLo
                v16bf b;
                #pragma unroll
                for (int e = 0; e < 16; ++e)
                    b[e] = sw[(nt * 16 + laneLo) * 64 + ks + (laneHi << 4) + e];
                acc[nt] = __builtin_amdgcn_wmma_f32_16x16x32_bf16(
                    false, a, false, b, (short)0, acc[nt], false, false);
            }
        }
        __syncthreads();
    }

    // store: acc VGPR r -> M = r + laneHi*8, N = laneLo
    #pragma unroll
    for (int nt = 0; nt < 8; ++nt) {
        #pragma unroll
        for (int r = 0; r < 8; ++r) {
            int gr = rowBase + wv * 16 + r + laneHi * 8;
            if (gr < N_NODES)
                h[(long long)gr * M_OUT + nt * 16 + laneLo] = acc[nt][r];
        }
    }
}

// ---------------------------------------------------------------------------
// float atomic max via signed/unsigned int trick (init must be -inf bits)
// ---------------------------------------------------------------------------
__device__ inline void atomicMaxFloat(float* addr, float val) {
    if (val >= 0.0f) atomicMax((int*)addr, __float_as_int(val));
    else             atomicMin((unsigned int*)addr, __float_as_uint(val));
}

// ---------------------------------------------------------------------------
// Pass 1: one wave32 per edge. lane = d (OUT_DIM==32), 4 heads unrolled.
// e[h] = tanh(<hs,a_src> + <hd,a_dst>) * w ; store + segment-max into emax.
// ---------------------------------------------------------------------------
__global__ __launch_bounds__(256)
void k_edge_scores(const float* __restrict__ h, const int* __restrict__ adj,
                   const float* __restrict__ ew, const float* __restrict__ a_src,
                   const float* __restrict__ a_dst, float* __restrict__ ebuf,
                   float* __restrict__ emax) {
    long long gid  = (long long)blockIdx.x * blockDim.x + threadIdx.x;
    int       lane = threadIdx.x & 31;
    long long edge = gid >> 5;
    if (edge >= N_EDGES) return;

    // prefetch upcoming edge records (emits global_prefetch_b8)
    __builtin_prefetch((const void*)(adj + 2 * edge + 4096), 0, 0);

    int src = adj[2 * edge + 0];
    int dst = adj[2 * edge + 1];
    const float* hs = h + (long long)src * M_OUT;
    const float* hd = h + (long long)dst * M_OUT;

    float p[HEADS];
    #pragma unroll
    for (int hh = 0; hh < HEADS; ++hh) {
        int idx = hh * OUT_DIM + lane;
        p[hh] = hs[idx] * a_src[idx] + hd[idx] * a_dst[idx];
    }
    #pragma unroll
    for (int hh = 0; hh < HEADS; ++hh) {
        #pragma unroll
        for (int off = 16; off >= 1; off >>= 1)
            p[hh] += __shfl_xor(p[hh], off, 32);
    }
    float w = ew[edge];
    if (lane == 0) {
        #pragma unroll
        for (int hh = 0; hh < HEADS; ++hh) {
            float e = tanhf(p[hh]) * w;
            ebuf[edge * HEADS + hh] = e;
            atomicMaxFloat(&emax[(long long)dst * HEADS + hh], e);
        }
    }
}

// ---------------------------------------------------------------------------
// emax fixup: non-finite (no incoming edges -> -inf) becomes 0
// ---------------------------------------------------------------------------
__global__ __launch_bounds__(256)
void k_fixmax(float* __restrict__ emax) {
    long long i = (long long)blockIdx.x * blockDim.x + threadIdx.x;
    if (i >= (long long)N_NODES * HEADS) return;
    float v = emax[i];
    if (!isfinite(v)) emax[i] = 0.0f;
}

// ---------------------------------------------------------------------------
// Pass 2: ex = exp(e - emax[dst]); denom[dst] += ex. One thread per edge.
// ---------------------------------------------------------------------------
__global__ __launch_bounds__(256)
void k_edge_exp(const int* __restrict__ adj, float* __restrict__ ebuf,
                const float* __restrict__ emax, float* __restrict__ denom) {
    long long i = (long long)blockIdx.x * blockDim.x + threadIdx.x;
    if (i >= N_EDGES) return;
    int dst = adj[2 * i + 1];
    #pragma unroll
    for (int hh = 0; hh < HEADS; ++hh) {
        float ex = expf(ebuf[i * HEADS + hh] - emax[(long long)dst * HEADS + hh]);
        ebuf[i * HEADS + hh] = ex;
        atomicAdd(&denom[(long long)dst * HEADS + hh], ex);
    }
}

// ---------------------------------------------------------------------------
// Pass 3: out[dst] += alpha * hs. One wave per edge; lane = d within head.
// Gathers hit L2 (h = 25.6 MB << 192 MB L2); scatters are L2 f32 atomics.
// ---------------------------------------------------------------------------
__global__ __launch_bounds__(256)
void k_edge_scatter(const float* __restrict__ h, const int* __restrict__ adj,
                    const float* __restrict__ ebuf, const float* __restrict__ denom,
                    float* __restrict__ out) {
    long long gid  = (long long)blockIdx.x * blockDim.x + threadIdx.x;
    int       lane = threadIdx.x & 31;
    long long edge = gid >> 5;
    if (edge >= N_EDGES) return;

    int src = adj[2 * edge + 0];
    int dst = adj[2 * edge + 1];
    const float* hs = h + (long long)src * M_OUT;

    #pragma unroll
    for (int hh = 0; hh < HEADS; ++hh) {
        float alpha = ebuf[edge * HEADS + hh] / denom[(long long)dst * HEADS + hh];
        int idx = hh * OUT_DIM + lane;
        atomicAdd(&out[(long long)dst * M_OUT + idx], alpha * hs[idx]);
    }
}

// ---------------------------------------------------------------------------
// exact GELU: 0.5*x*(1+erf(x/sqrt(2)))
// ---------------------------------------------------------------------------
__global__ __launch_bounds__(256)
void k_gelu(float* __restrict__ out) {
    long long i = (long long)blockIdx.x * blockDim.x + threadIdx.x;
    if (i >= (long long)N_NODES * M_OUT) return;
    float v = out[i];
    out[i] = 0.5f * v * (1.0f + erff(v * 0.70710678118654752f));
}

// ---------------------------------------------------------------------------
extern "C" void kernel_launch(void* const* d_in, const int* in_sizes, int n_in,
                              void* d_out, int out_size, void* d_ws, size_t ws_size,
                              hipStream_t stream) {
    const float* x      = (const float*)d_in[0];
    const int*   adj    = (const int*)d_in[1];
    const float* ew     = (const float*)d_in[2];
    const float* W      = (const float*)d_in[3];
    const float* a_src  = (const float*)d_in[4];
    const float* a_dst  = (const float*)d_in[5];
    float* out = (float*)d_out;

    // workspace layout
    float* hbuf  = (float*)d_ws;                                   // N*128
    float* ebuf  = hbuf  + (size_t)N_NODES * M_OUT;                // E*4
    float* emax  = ebuf  + (size_t)N_EDGES * HEADS;                // N*4
    float* denom = emax  + (size_t)N_NODES * HEADS;                // N*4

    const long long outElems = (long long)N_NODES * M_OUT;         // 6.4M
    const int elemBlocks = (int)((outElems + 255) / 256);          // 25000
    const int gemmBlocks = (N_NODES + 127) / 128;                  // 391
    const long long waveThreads = (long long)N_EDGES * 32;
    const int edgeWaveBlocks = (int)((waveThreads + 255) / 256);   // 200000
    const int edgeBlocks = (N_EDGES + 255) / 256;                  // 6250
    const int nhBlocks = (int)(((long long)N_NODES * HEADS + 255) / 256);

    k_init        <<<elemBlocks,     256, 0, stream>>>(out, emax, denom);
    k_gemm_wmma   <<<gemmBlocks,     256, 0, stream>>>(x, W, hbuf);
    k_edge_scores <<<edgeWaveBlocks, 256, 0, stream>>>(hbuf, adj, ew, a_src, a_dst, ebuf, emax);
    k_fixmax      <<<nhBlocks,       256, 0, stream>>>(emax);
    k_edge_exp    <<<edgeBlocks,     256, 0, stream>>>(adj, ebuf, emax, denom);
    k_edge_scatter<<<edgeWaveBlocks, 256, 0, stream>>>(hbuf, adj, ebuf, denom, out);
    k_gelu        <<<elemBlocks,     256, 0, stream>>>(out);
}
